// AttentionHead_64939905515872
// MI455X (gfx1250) — matrix operand order repaired and checked by reading, not compiled, not run on hardware
//
#include <hip/hip_runtime.h>

// ---------------------------------------------------------------------------
// Causal attention head, B=4 T=4096 C=1024 D=64, fp32 IO.
// bf16 WMMA (v_wmma_f32_16x16x32_bf16, f32 accum) flash attention.
// K/V tiles staged in LDS by the Tensor Data Mover (tensor_load_to_lds,
// TENSORcnt double-buffered) and shared by 4 waves per block to cut L2
// read amplification 4x. Projections are software-pipelined WMMA.
// ---------------------------------------------------------------------------

typedef __bf16 bf16_t;
typedef __attribute__((ext_vector_type(16))) __bf16 v16bf;
typedef __attribute__((ext_vector_type(8)))  __bf16 v8bf;
typedef __attribute__((ext_vector_type(8)))  float  v8f;
typedef __attribute__((ext_vector_type(4)))  unsigned int v4u;
typedef __attribute__((ext_vector_type(8)))  int v8i;
typedef __attribute__((ext_vector_type(4)))  int v4i;

#define NB 4
#define NT 4096
#define NC 1024
#define ND 64

union V16U { v16bf v; v8bf h[2]; };

static __device__ __forceinline__ v8f zero8() {
  v8f z;
#pragma unroll
  for (int i = 0; i < 8; ++i) z[i] = 0.0f;
  return z;
}

static __device__ __forceinline__ v8f wmma_bf16(v16bf a, v16bf b, v8f c) {
  // 8 args: (neg_a, A, neg_b, B, c_mod, C, reuse_a, reuse_b)
  return __builtin_amdgcn_wmma_f32_16x16x32_bf16(false, a, false, b,
                                                 (short)0, c, false, false);
}

// Low 32 bits of a flat shared-aperture address = logical LDS byte offset.
static __device__ __forceinline__ unsigned lds_off(const void* p) {
  return (unsigned)(uintptr_t)p;
}

// TDM: 2D tile load Global -> LDS, descriptor per CDNA5 ISA ch.8.
// dims/strides in elements (data_size = 2 bytes).
// This toolchain (clang-23 / therock) exposes the 6-arg builtin:
//   (uint32x4 g0, int32x8 g1, int32x4 g2, int32x4 g3, int32x8 extra, i32 cpol)
// Groups 2/3 and the extra group are unused for a 2D tile -> zero-filled.
static __device__ __forceinline__ void tdm_load_2d(
    unsigned lds_byte_off, const void* gptr,
    unsigned tensor_d0, unsigned tensor_d1,
    unsigned tile_d0, unsigned tile_d1,
    unsigned long long stride0) {
  unsigned long long ga = (unsigned long long)(uintptr_t)gptr;
  v4u g0;
  g0[0] = 1u;                                   // count=1 valid user D#
  g0[1] = lds_byte_off;                         // lds_addr
  g0[2] = (unsigned)ga;                         // global_addr[31:0]
  g0[3] = (unsigned)((ga >> 32) & 0x01ffffffu) | (2u << 30);  // addr[56:32]|type=2
  v8i g1;
  g1[0] = (int)(1u << 16);                      // workgroup_mask=0, data_size=1 (2B)
  g1[1] = (int)((tensor_d0 & 0xffffu) << 16);   // tensor_dim0[15:0]
  g1[2] = (int)(((tensor_d0 >> 16) & 0xffffu) | ((tensor_d1 & 0xffffu) << 16));
  g1[3] = (int)(((tensor_d1 >> 16) & 0xffffu) | ((tile_d0 & 0xffffu) << 16));
  g1[4] = (int)(tile_d1 & 0xffffu);             // tile_dim1 ; tile_dim2=0
  g1[5] = (int)(unsigned)(stride0 & 0xffffffffu);        // dim0_stride[31:0]
  g1[6] = (int)(unsigned)((stride0 >> 32) & 0xffffu);    // dim0_stride[47:32]
  g1[7] = 0;                                    // dim1_stride (unused, 2D)
  v4i g2, g3;
#pragma unroll
  for (int i = 0; i < 4; ++i) { g2[i] = 0; g3[i] = 0; }
  v8i gx;
#pragma unroll
  for (int i = 0; i < 8; ++i) gx[i] = 0;
  __builtin_amdgcn_tensor_load_to_lds(g0, g1, g2, g3, gx, 0);
}

// ---------------------------------------------------------------------------
// Kernel A: W[1024][64] f32 -> Wt[64][1024] bf16 (transposed for B-fragments).
// ---------------------------------------------------------------------------
__global__ void __launch_bounds__(256) wconv_kernel(
    const float* __restrict__ WQ, const float* __restrict__ WV,
    bf16_t* __restrict__ WtQ, bf16_t* __restrict__ WtV) {
  int i = blockIdx.x * blockDim.x + threadIdx.x;   // 0 .. 65535
  int c = i >> 6;
  int d = i & 63;
  WtQ[(size_t)d * NC + c] = (bf16_t)WQ[i];
  WtV[(size_t)d * NC + c] = (bf16_t)WV[i];
}

// ---------------------------------------------------------------------------
// Kernel B: projections, one wave per 16 rows, software-pipelined k-loop.
// Q = x@WQ ; K(=V) = x@WV. Stores Qb/Kb row-major bf16 and Vt[B][64][T].
// ---------------------------------------------------------------------------
__global__ void __launch_bounds__(128) proj_kernel(
    const float* __restrict__ x,
    const bf16_t* __restrict__ WtQ, const bf16_t* __restrict__ WtV,
    bf16_t* __restrict__ Qb, bf16_t* __restrict__ Kb,
    bf16_t* __restrict__ Vt) {
  const int wave = blockIdx.x * (blockDim.x >> 5) + (threadIdx.x >> 5);
  const int lane = threadIdx.x & 31;
  const int n = lane & 15;
  const int h = lane >> 4;
  const int row0 = wave * 16;

  v8f accQ[4], accK[4];
#pragma unroll
  for (int i = 0; i < 4; ++i) { accQ[i] = zero8(); accK[i] = zero8(); }

  const float* xrow = x + (size_t)(row0 + n) * NC;

  auto loadA = [&](int kc, V16U& a) {
    const float4* xp = (const float4*)(xrow + kc + h * 16);
    const float4* xq = (const float4*)(xrow + kc + h * 16 + 8);
    float4 f0 = xp[0], f1 = xp[1], f2 = xq[0], f3 = xq[1];
    v8bf lo, hi;
    lo[0] = (bf16_t)f0.x; lo[1] = (bf16_t)f0.y; lo[2] = (bf16_t)f0.z; lo[3] = (bf16_t)f0.w;
    lo[4] = (bf16_t)f1.x; lo[5] = (bf16_t)f1.y; lo[6] = (bf16_t)f1.z; lo[7] = (bf16_t)f1.w;
    hi[0] = (bf16_t)f2.x; hi[1] = (bf16_t)f2.y; hi[2] = (bf16_t)f2.z; hi[3] = (bf16_t)f2.w;
    hi[4] = (bf16_t)f3.x; hi[5] = (bf16_t)f3.y; hi[6] = (bf16_t)f3.z; hi[7] = (bf16_t)f3.w;
    a.h[0] = lo; a.h[1] = hi;
  };
  auto loadB = [&](int kc, v16bf* bq, v16bf* bv) {
#pragma unroll
    for (int cb = 0; cb < 4; ++cb) {
      bq[cb] = *(const v16bf*)(WtQ + (size_t)(cb * 16 + n) * NC + kc + h * 16);
      bv[cb] = *(const v16bf*)(WtV + (size_t)(cb * 16 + n) * NC + kc + h * 16);
    }
  };

  V16U a;
  v16bf bq[4], bv[4];
  loadA(0, a);
  loadB(0, bq, bv);
  for (int kc = 32; kc < NC; kc += 32) {
    V16U a2; v16bf bq2[4], bv2[4];
    loadA(kc, a2);            // next iteration's fragments issue first,
    loadB(kc, bq2, bv2);      // overlapping the WMMAs below
#pragma unroll
    for (int cb = 0; cb < 4; ++cb) {
      accQ[cb] = wmma_bf16(a.v, bq[cb], accQ[cb]);
      accK[cb] = wmma_bf16(a.v, bv[cb], accK[cb]);
    }
    a = a2;
#pragma unroll
    for (int cb = 0; cb < 4; ++cb) { bq[cb] = bq2[cb]; bv[cb] = bv2[cb]; }
  }
#pragma unroll
  for (int cb = 0; cb < 4; ++cb) {
    accQ[cb] = wmma_bf16(a.v, bq[cb], accQ[cb]);
    accK[cb] = wmma_bf16(a.v, bv[cb], accK[cb]);
  }

#pragma unroll
  for (int cb = 0; cb < 4; ++cb) {
#pragma unroll
    for (int r = 0; r < 8; ++r) {
      int ro = row0 + r + 8 * h;
      int col = cb * 16 + n;
      bf16_t qv = (bf16_t)accQ[cb][r];
      bf16_t kv = (bf16_t)accK[cb][r];
      Qb[(size_t)ro * ND + col] = qv;
      Kb[(size_t)ro * ND + col] = kv;
      int bb = ro >> 12;
      int tt = ro & (NT - 1);
      Vt[((size_t)bb * ND + col) * NT + tt] = kv;
    }
  }
}

// ---------------------------------------------------------------------------
// Kernel C: flash attention. 1 block = 4 waves = 64 query rows. K/V tiles
// (32 keys) TDM-loaded into double-buffered LDS by wave 0, consumed by all
// 4 waves. Online softmax with width-16 shfl reductions; P reswizzled via
// per-wave LDS tile; O += P@V accumulated by WMMA.
// ---------------------------------------------------------------------------
#define WPB 4
__global__ void __launch_bounds__(128) attn_kernel(
    const bf16_t* __restrict__ Qb, const bf16_t* __restrict__ Kb,
    const bf16_t* __restrict__ Vt, float* __restrict__ out) {
  __shared__ __align__(32) bf16_t ldsK[2][32 * 64];   // [key][d]
  __shared__ __align__(32) bf16_t ldsV[2][64 * 32];   // [d-col][key]
  __shared__ __align__(32) bf16_t ldsP[WPB][16 * 32];

  const int w = threadIdx.x >> 5;
  const int lane = threadIdx.x & 31;
  const int n = lane & 15;
  const int h = lane >> 4;
  const int bb = blockIdx.x >> 6;          // 64 blocks per batch
  const int rb = (blockIdx.x & 63) * 64;   // block's query-row base
  const int trow = rb + w * 16;            // this wave's query tile
  const size_t base = (size_t)bb * NT;
  const int jmax = rb + 64;                // keys needed by the block

  // Q A-fragments (d=0..31, 32..63), loaded once.
  const bf16_t* qrow = Qb + (base + trow + n) * ND;
  V16U aq0, aq1;
  aq0.h[0] = *(const v8bf*)(qrow + h * 16);
  aq0.h[1] = *(const v8bf*)(qrow + h * 16 + 8);
  aq1.h[0] = *(const v8bf*)(qrow + 32 + h * 16);
  aq1.h[1] = *(const v8bf*)(qrow + 40 + h * 16);

  v8f o[4];
  float m[8], l[8];
#pragma unroll
  for (int cb = 0; cb < 4; ++cb) o[cb] = zero8();
#pragma unroll
  for (int r = 0; r < 8; ++r) { m[r] = -3.0e38f; l[r] = 0.0f; }

  bf16_t* pt = ldsP[w];
  const bf16_t* vbase = Vt + (size_t)bb * ND * NT;

  // Prologue: TDM-load K/V tile for j=0 into buffer 0.
  if (w == 0) {
    tdm_load_2d(lds_off(ldsK[0]), Kb + base * ND, ND, NT, ND, 32, ND);
    tdm_load_2d(lds_off(ldsV[0]), vbase, NT, ND, 32, ND, NT);
  }

  int it = 0;
  for (int j = 0; j < jmax; j += 32, ++it) {
    const int p = it & 1;
    if (w == 0) {
      int jn = j + 32;
      if (jn < jmax) {
        tdm_load_2d(lds_off(ldsK[1 - p]), Kb + (base + jn) * ND, ND, NT, ND, 32, ND);
        tdm_load_2d(lds_off(ldsV[1 - p]), vbase + jn, NT, ND, 32, ND, NT);
        __builtin_amdgcn_s_wait_tensorcnt(2);  // buf[p]'s two loads done
      } else {
        __builtin_amdgcn_s_wait_tensorcnt(0);
      }
    }
    __syncthreads();   // buf[p] visible to all waves

    if (j < trow + 16) {   // uniform per wave: causal range
      const bf16_t* kp = ldsK[p];
      const bf16_t* vp = ldsV[p];
      v16bf b00 = *(const v16bf*)(kp + (size_t)n * ND + h * 16);
      v16bf b01 = *(const v16bf*)(kp + (size_t)n * ND + 32 + h * 16);
      v16bf b10 = *(const v16bf*)(kp + (size_t)(16 + n) * ND + h * 16);
      v16bf b11 = *(const v16bf*)(kp + (size_t)(16 + n) * ND + 32 + h * 16);

      v8f c0 = zero8(), c1 = zero8();
      c0 = wmma_bf16(aq0.v, b00, c0);
      c0 = wmma_bf16(aq1.v, b01, c0);
      c1 = wmma_bf16(aq0.v, b10, c1);
      c1 = wmma_bf16(aq1.v, b11, c1);

#pragma unroll
      for (int r = 0; r < 8; ++r) {
        int qi = trow + r + 8 * h;
        float s0 = (j + n      <= qi) ? c0[r] : -1e30f;
        float s1 = (j + 16 + n <= qi) ? c1[r] : -1e30f;
        float mx = fmaxf(s0, s1);
#pragma unroll
        for (int d = 1; d < 16; d <<= 1) mx = fmaxf(mx, __shfl_xor(mx, d, 16));
        float mn = fmaxf(m[r], mx);
        float sc = __expf(m[r] - mn);
        m[r] = mn;
        float p0 = __expf(s0 - mn);
        float p1 = __expf(s1 - mn);
        float rs = p0 + p1;
#pragma unroll
        for (int d = 1; d < 16; d <<= 1) rs += __shfl_xor(rs, d, 16);
        l[r] = l[r] * sc + rs;
#pragma unroll
        for (int cb = 0; cb < 4; ++cb) o[cb][r] *= sc;
        pt[(r + 8 * h) * 32 + n]      = (bf16_t)p0;
        pt[(r + 8 * h) * 32 + 16 + n] = (bf16_t)p1;
      }
      __asm__ volatile("" ::: "memory");   // keep per-wave LDS store->load order

      V16U ap;
      ap.h[0] = *(const v8bf*)(pt + n * 32 + h * 16);
      ap.h[1] = *(const v8bf*)(pt + n * 32 + h * 16 + 8);

#pragma unroll
      for (int cb = 0; cb < 4; ++cb) {
        v16bf bvf = *(const v16bf*)(vp + (size_t)(cb * 16 + n) * 32 + h * 16);
        o[cb] = wmma_bf16(ap.v, bvf, o[cb]);
      }
    }
    __syncthreads();   // all waves done with buf[p] before it is reloaded
  }

#pragma unroll
  for (int r = 0; r < 8; ++r) {
    float inv = 1.0f / l[r];
    size_t ro = base + trow + r + 8 * h;
#pragma unroll
    for (int cb = 0; cb < 4; ++cb)
      out[ro * ND + cb * 16 + n] = o[cb][r] * inv;
  }
}

// ---------------------------------------------------------------------------
extern "C" void kernel_launch(void* const* d_in, const int* in_sizes, int n_in,
                              void* d_out, int out_size, void* d_ws, size_t ws_size,
                              hipStream_t stream) {
  const float* x  = (const float*)d_in[0];
  const float* WQ = (const float*)d_in[1];
  // d_in[2] = WK: present but unused in the reference math (source bug k = x@WV).
  const float* WV = (const float*)d_in[3];
  float* out = (float*)d_out;

  bf16_t* WtQ = (bf16_t*)d_ws;                 // 128 KB
  bf16_t* WtV = WtQ + (size_t)ND * NC;         // 128 KB
  bf16_t* Qb  = WtV + (size_t)ND * NC;         // 2 MB
  bf16_t* Kb  = Qb  + (size_t)NB * NT * ND;    // 2 MB
  bf16_t* Vt  = Kb  + (size_t)NB * NT * ND;    // 2 MB  (total ~6.25 MB)

  wconv_kernel<<<256, 256, 0, stream>>>(WQ, WV, WtQ, WtV);
  proj_kernel<<<256, 128, 0, stream>>>(x, WtQ, WtV, Qb, Kb, Vt);   // 1024 waves
  attn_kernel<<<256, 128, 0, stream>>>(Qb, Kb, Vt, out);           // 256 blocks x 4 waves
}
